// SimpleAbsorbedAttention_58531814310127
// MI455X (gfx1250) — matrix operand order repaired and checked by reading, not compiled, not run on hardware
//
#include <hip/hip_runtime.h>
#include <hip/hip_bf16.h>

// MLA absorbed attention for MI455X (gfx1250, wave32, WMMA bf16 16x16x32).
// Heavy GEMMs done in bf16 with f32 accumulation -> bandwidth-bound
// (~420MB HBM traffic @ 23.3TB/s). compressed_kv streamed once per
// (b, head-group) with flash-style online softmax.

#define B_   32
#define S_   4096
#define H_   128
#define DH_  128
#define R_   512
#define QR_  1536

typedef __bf16 bf16x16 __attribute__((ext_vector_type(16)));
typedef float  f32x8   __attribute__((ext_vector_type(8)));

union FragAB { unsigned int u[8]; bf16x16 v; };

__device__ __forceinline__ unsigned short f2bf(float x) {
  unsigned int u = __float_as_uint(x);
  u += 0x7FFFu + ((u >> 16) & 1u);            // round to nearest even
  return (unsigned short)(u >> 16);
}
__device__ __forceinline__ unsigned int pack2(float lo, float hi) {
  return (unsigned int)f2bf(lo) | ((unsigned int)f2bf(hi) << 16);
}

// 16-bit A-matrix 16x32 layout (ISA 7.12.2): lane m=lane&15,
// v0..3: K = 2(v&3) (+8 for lanes>=16), v4..7: +16.
__device__ __forceinline__ int koffA(int v, int hi) { return (v & 3) * 2 + (v >> 2) * 16 + hi * 8; }
// B-matrix 32x16: lane n=lane&15, same K order as A: v0..3: K = 2(v&3) (+8 for lanes>=16), v4..7: +16.
__device__ __forceinline__ int koffB(int v, int hi) { return (v & 3) * 2 + (v >> 2) * 16 + hi * 8; }

__device__ __forceinline__ f32x8 wmma_bf16(const FragAB& a, const FragAB& b, f32x8 c) {
  return __builtin_amdgcn_wmma_f32_16x16x32_bf16(false, a.v, false, b.v, (short)0, c, false, false);
}

// A fragment from LDS: rows m (16 of them) with row stride ld (bf16 elems), k chunk at k0.
__device__ __forceinline__ void loadA(FragAB& f, const unsigned short* p, int ld, int k0) {
  const int lane = threadIdx.x & 31;
  const int hi = lane >> 4, m = lane & 15;
#pragma unroll
  for (int v = 0; v < 8; ++v)
    f.u[v] = *(const unsigned int*)(p + m * ld + k0 + koffA(v, hi));
}
// B fragment from LDS stored as [n][k] (n-major, row stride ld), k chunk at k0.
__device__ __forceinline__ void loadB(FragAB& f, const unsigned short* p, int ld, int k0) {
  const int lane = threadIdx.x & 31;
  const int hi = lane >> 4, n = lane & 15;
#pragma unroll
  for (int v = 0; v < 8; ++v)
    f.u[v] = *(const unsigned int*)(p + n * ld + k0 + koffB(v, hi));
}

// ---------------- Kernel 1: q = hq @ wq^T + bq  ->  q_bf [32][16384] bf16 ----------------
// grid 64, block 128 (4 waves). Each WG: 256 output cols, M=32 (2 m-tiles), K=1536.
__global__ void k_qproj(const float* __restrict__ hq, const float* __restrict__ wq,
                        const float* __restrict__ bq, unsigned short* __restrict__ q_bf) {
  extern __shared__ char smem[];
  unsigned short* aL = (unsigned short*)smem;   // [32][64]
  unsigned short* bL = aL + 32 * 64;            // [256][64]  ([n][k])
  const int tid = threadIdx.x;
  const int wave = tid >> 5, lane = tid & 31;
  const int hi = lane >> 4, ln = lane & 15;
  const int nBase = blockIdx.x * 256;

  f32x8 acc[2][4];
#pragma unroll
  for (int mi = 0; mi < 2; ++mi)
#pragma unroll
    for (int ni = 0; ni < 4; ++ni) acc[mi][ni] = (f32x8)0.f;

  for (int kc = 0; kc < QR_ / 64; ++kc) {
    const int k0 = kc * 64;
    for (int i = 0; i < 8; ++i) {               // A: 32x64 f32 -> 1024 bf16 pairs
      int p = i * 128 + tid, e = p * 2, m = e >> 6, k = e & 63;
      const float* s = hq + (size_t)m * QR_ + k0 + k;
      ((unsigned int*)aL)[p] = pack2(s[0], s[1]);
    }
    for (int i = 0; i < 64; ++i) {              // B: 256x64 f32 -> 8192 pairs (wq is [n][k])
      int p = i * 128 + tid, e = p * 2, n = e >> 6, k = e & 63;
      const float* s = wq + (size_t)(nBase + n) * QR_ + k0 + k;
      ((unsigned int*)bL)[p] = pack2(s[0], s[1]);
    }
    __syncthreads();
#pragma unroll
    for (int kk = 0; kk < 2; ++kk) {
      FragAB A0, A1;
      loadA(A0, aL, 64, kk * 32);
      loadA(A1, aL + 16 * 64, 64, kk * 32);
#pragma unroll
      for (int ni = 0; ni < 4; ++ni) {
        FragAB Bf;
        loadB(Bf, bL + (wave * 4 + ni) * 16 * 64, 64, kk * 32);
        acc[0][ni] = wmma_bf16(A0, Bf, acc[0][ni]);
        acc[1][ni] = wmma_bf16(A1, Bf, acc[1][ni]);
      }
    }
    __syncthreads();
  }
#pragma unroll
  for (int ni = 0; ni < 4; ++ni) {
    int nl = (wave * 4 + ni) * 16 + ln;
    float bias = bq[nBase + nl];
#pragma unroll
    for (int mi = 0; mi < 2; ++mi)
#pragma unroll
      for (int v = 0; v < 8; ++v) {
        int m = mi * 16 + v + 8 * hi;
        q_bf[(size_t)m * (H_ * DH_) + nBase + nl] = f2bf(acc[mi][ni][v] + bias);
      }
  }
}

// ------------- Kernel 2: qa[b,h,r] = sum_d q[b,h,d]*w_kc[h,d,r] (per-h GEMM) -------------
// grid 128 (h), block 256 (8 waves). M=32(b), N=512(r), K=128(d).
__global__ void k_qabsorb(const unsigned short* __restrict__ q_bf, const float* __restrict__ w_kc,
                          unsigned short* __restrict__ qa_bf) {
  extern __shared__ char smem[];
  unsigned short* aL  = (unsigned short*)smem;  // [32][128]  (b-major)
  unsigned short* wtL = aL + 32 * 128;          // [512][128] (r-major, d contiguous)
  const int tid = threadIdx.x;
  const int wave = tid >> 5, lane = tid & 31;
  const int hi = lane >> 4, ln = lane & 15;
  const int h = blockIdx.x;

  for (int i = 0; i < 8; ++i) {                 // copy 32x128 bf16 (already bf16 in ws)
    int p = i * 256 + tid, e = p * 2, m = e >> 7, d = e & 127;
    ((unsigned int*)aL)[p] = *(const unsigned int*)(q_bf + (size_t)m * (H_ * DH_) + h * DH_ + d);
  }
  const float* wsrc = w_kc + (size_t)h * DH_ * R_;
  for (int i = 0; i < 256; ++i) {               // transpose-stage w_kc[h]: [d][r] f32 -> [r][d] bf16
    int f = i * 256 + tid, d = f >> 9, r = f & 511;
    wtL[r * 128 + d] = f2bf(wsrc[f]);
  }
  __syncthreads();

  f32x8 acc[2][4];
#pragma unroll
  for (int mi = 0; mi < 2; ++mi)
#pragma unroll
    for (int ni = 0; ni < 4; ++ni) acc[mi][ni] = (f32x8)0.f;
#pragma unroll
  for (int kc = 0; kc < 4; ++kc) {
    FragAB A0, A1;
    loadA(A0, aL, 128, kc * 32);
    loadA(A1, aL + 16 * 128, 128, kc * 32);
#pragma unroll
    for (int ni = 0; ni < 4; ++ni) {
      FragAB Bf;
      loadB(Bf, wtL + (wave * 64 + ni * 16) * 128, 128, kc * 32);
      acc[0][ni] = wmma_bf16(A0, Bf, acc[0][ni]);
      acc[1][ni] = wmma_bf16(A1, Bf, acc[1][ni]);
    }
  }
#pragma unroll
  for (int ni = 0; ni < 4; ++ni) {
    int r = wave * 64 + ni * 16 + ln;
#pragma unroll
    for (int mi = 0; mi < 2; ++mi)
#pragma unroll
      for (int v = 0; v < 8; ++v) {
        int b = mi * 16 + v + 8 * hi;
        qa_bf[((size_t)b * H_ + h) * R_ + r] = f2bf(acc[mi][ni][v]);
      }
  }
}

// -------- Kernel 3: fused attention (scores -> online softmax -> context), per (b, 16 heads) --------
// grid (8, 32), block 128 (4 waves). S_BLK=64. Wave w: scores s-tile w; context r-range [w*128, w*128+128).
__global__ void k_attn(const unsigned short* __restrict__ qa_bf, const float* __restrict__ ckv,
                       unsigned short* __restrict__ ctx_bf) {
  extern __shared__ char smem[];
  unsigned short* kvL = (unsigned short*)smem;   // [64][512]  (s-major: scores B)
  unsigned short* kvT = kvL + 64 * 512;          // [512][64]  (r-major: context B)
  unsigned short* qaL = kvT + 512 * 64;          // [16][512]
  unsigned short* prL = qaL + 16 * 512;          // [16][64]   probs bf16
  float* blkmax = (float*)(prL + 16 * 64);       // [4][16]
  float* blksum = blkmax + 64;                   // [4][16]
  float* runmax = blksum + 64;                   // [16]
  float* runsum = runmax + 16;                   // [16]
  float* scaleL = runsum + 16;                   // [16]
  const int tid = threadIdx.x;
  const int wave = tid >> 5, lane = tid & 31;
  const int hi = lane >> 4, ln = lane & 15;
  const int hg = blockIdx.x, b = blockIdx.y;
  const int h0 = hg * 16;

  const unsigned int* qsrc = (const unsigned int*)(qa_bf + ((size_t)b * H_ + h0) * R_);
  for (int i = 0; i < 32; ++i) ((unsigned int*)qaL)[i * 128 + tid] = qsrc[i * 128 + tid];
  if (tid < 16) { runmax[tid] = -3.0e38f; runsum[tid] = 0.f; }

  f32x8 ctx[8];
#pragma unroll
  for (int t = 0; t < 8; ++t) ctx[t] = (f32x8)0.f;
  __syncthreads();

  const float* kvbase = ckv + (size_t)b * S_ * R_;
  for (int sb = 0; sb < S_ / 64; ++sb) {
    const int s0 = sb * 64;
    for (int i = 0; i < 128; ++i) {              // stage 64x512 f32 block -> bf16 (both layouts)
      int p = i * 128 + tid, e = p * 2, s = e >> 9, r = e & 511;
      const float* src = kvbase + (size_t)(s0 + s) * R_ + r;
      unsigned short blo = f2bf(src[0]), bhi = f2bf(src[1]);
      ((unsigned int*)kvL)[p] = (unsigned)blo | ((unsigned)bhi << 16);
      kvT[(size_t)r * 64 + s] = blo;
      kvT[(size_t)(r + 1) * 64 + s] = bhi;
    }
    __syncthreads();

    // scores tile [16h x 16s] for this wave, K = 512
    f32x8 sc = (f32x8)0.f;
#pragma unroll
    for (int kc = 0; kc < 16; ++kc) {
      FragAB A, Bf;
      loadA(A, qaL, 512, kc * 32);
      loadB(Bf, kvL + (wave * 16) * 512, 512, kc * 32);
      sc = wmma_bf16(A, Bf, sc);
    }

    // per-wave row max over its 16 columns (shfl within lane halves)
#pragma unroll
    for (int v = 0; v < 8; ++v) {
      float x = sc[v];
      x = fmaxf(x, __shfl_xor(x, 1, 32));
      x = fmaxf(x, __shfl_xor(x, 2, 32));
      x = fmaxf(x, __shfl_xor(x, 4, 32));
      x = fmaxf(x, __shfl_xor(x, 8, 32));
      if (ln == 0) blkmax[wave * 16 + v + 8 * hi] = x;
    }
    __syncthreads();
    if (tid < 16) {                              // online-softmax state update (per head row)
      float m = fmaxf(fmaxf(blkmax[tid], blkmax[16 + tid]), fmaxf(blkmax[32 + tid], blkmax[48 + tid]));
      float mOld = runmax[tid], mNew = fmaxf(mOld, m);
      float s = __expf(mOld - mNew);
      runmax[tid] = mNew; scaleL[tid] = s; runsum[tid] *= s;
    }
    __syncthreads();

    float rm2[8], scv[8], ps[8];
#pragma unroll
    for (int v = 0; v < 8; ++v) { rm2[v] = runmax[v + 8 * hi]; scv[v] = scaleL[v + 8 * hi]; }
#pragma unroll
    for (int v = 0; v < 8; ++v) {
      float p = __expf(sc[v] - rm2[v]);
      prL[(v + 8 * hi) * 64 + wave * 16 + ln] = f2bf(p);
      float s = p;
      s += __shfl_xor(s, 1, 32);
      s += __shfl_xor(s, 2, 32);
      s += __shfl_xor(s, 4, 32);
      s += __shfl_xor(s, 8, 32);
      ps[v] = s;
    }
    if (ln == 0)
#pragma unroll
      for (int v = 0; v < 8; ++v) blksum[wave * 16 + v + 8 * hi] = ps[v];
#pragma unroll
    for (int t = 0; t < 8; ++t)
#pragma unroll
      for (int v = 0; v < 8; ++v) ctx[t][v] *= scv[v];
    __syncthreads();
    if (tid < 16)                                // deterministic cross-wave sum accumulation
      runsum[tid] += ((blksum[tid] + blksum[16 + tid]) + (blksum[32 + tid] + blksum[48 + tid]));

    // context accumulation: A = probs [16h x 64s], B = kvT tiles, K = 64
#pragma unroll
    for (int kc = 0; kc < 2; ++kc) {
      FragAB A;
      loadA(A, prL, 64, kc * 32);
#pragma unroll
      for (int t = 0; t < 8; ++t) {
        FragAB Bf;
        loadB(Bf, kvT + (size_t)(wave * 128 + t * 16) * 64, 64, kc * 32);
        ctx[t] = wmma_bf16(A, Bf, ctx[t]);
      }
    }
    __syncthreads();
  }

  float inv[8];
#pragma unroll
  for (int v = 0; v < 8; ++v) inv[v] = 1.0f / runsum[v + 8 * hi];
#pragma unroll
  for (int t = 0; t < 8; ++t) {
    int r = wave * 128 + t * 16 + ln;
#pragma unroll
    for (int v = 0; v < 8; ++v) {
      int h = h0 + v + 8 * hi;
      ctx_bf[((size_t)b * H_ + h) * R_ + r] = f2bf(ctx[t][v] * inv[v]);
    }
  }
}

// ---------- Kernel 4: out[b,h,d] = sum_r w_vc[h,d,r]*ctx[b,h,r] (per-h GEMM, f32 out) ----------
// grid 128 (h), block 128 (4 waves). M=32(b), N=128(d), K=512(r).
__global__ void k_outproj(const unsigned short* __restrict__ ctx_bf, const float* __restrict__ w_vc,
                          float* __restrict__ out) {
  extern __shared__ char smem[];
  unsigned short* aL = (unsigned short*)smem;   // [32][512]
  unsigned short* bL = aL + 32 * 512;           // [128][512] ([d][r] — w_vc native layout)
  const int tid = threadIdx.x;
  const int wave = tid >> 5, lane = tid & 31;
  const int hi = lane >> 4, ln = lane & 15;
  const int h = blockIdx.x;

  for (int i = 0; i < 64; ++i) {
    int p = i * 128 + tid, e = p * 2, m = e >> 9, r = e & 511;
    ((unsigned int*)aL)[p] = *(const unsigned int*)(ctx_bf + ((size_t)m * H_ + h) * R_ + r);
  }
  const float* wsrc = w_vc + (size_t)h * DH_ * R_;
  for (int i = 0; i < 256; ++i) {
    int p = i * 128 + tid;
    const float* s = wsrc + (size_t)p * 2;
    ((unsigned int*)bL)[p] = pack2(s[0], s[1]);
  }
  __syncthreads();

  f32x8 acc[2][2];
#pragma unroll
  for (int mi = 0; mi < 2; ++mi)
#pragma unroll
    for (int ni = 0; ni < 2; ++ni) acc[mi][ni] = (f32x8)0.f;
#pragma unroll
  for (int kc = 0; kc < 16; ++kc) {
    FragAB A0, A1;
    loadA(A0, aL, 512, kc * 32);
    loadA(A1, aL + 16 * 512, 512, kc * 32);
#pragma unroll
    for (int ni = 0; ni < 2; ++ni) {
      FragAB Bf;
      loadB(Bf, bL + (wave * 32 + ni * 16) * 512, 512, kc * 32);
      acc[0][ni] = wmma_bf16(A0, Bf, acc[0][ni]);
      acc[1][ni] = wmma_bf16(A1, Bf, acc[1][ni]);
    }
  }
#pragma unroll
  for (int ni = 0; ni < 2; ++ni) {
    int d = wave * 32 + ni * 16 + ln;
#pragma unroll
    for (int mi = 0; mi < 2; ++mi)
#pragma unroll
      for (int v = 0; v < 8; ++v) {
        int bb = mi * 16 + v + 8 * hi;
        out[(size_t)bb * (H_ * DH_) + h * DH_ + d] = acc[mi][ni][v];
      }
  }
}

extern "C" void kernel_launch(void* const* d_in, const int* in_sizes, int n_in,
                              void* d_out, int out_size, void* d_ws, size_t ws_size,
                              hipStream_t stream) {
  (void)in_sizes; (void)n_in; (void)out_size; (void)ws_size;
  const float* hq   = (const float*)d_in[0];
  const float* ckv  = (const float*)d_in[1];
  const float* wq   = (const float*)d_in[2];
  const float* bq   = (const float*)d_in[3];
  const float* w_kc = (const float*)d_in[4];
  const float* w_vc = (const float*)d_in[5];
  float* out = (float*)d_out;

  // workspace: q_bf (1MB) | qa_bf (4MB) | ctx_bf (4MB) — ~9.4MB total
  unsigned short* q_bf   = (unsigned short*)d_ws;
  unsigned short* qa_bf  = q_bf + (size_t)B_ * H_ * DH_;
  unsigned short* ctx_bf = qa_bf + (size_t)B_ * H_ * R_;

  size_t sh1 = (size_t)(32 * 64 + 256 * 64) * 2;
  size_t sh2 = (size_t)(32 * 128 + 512 * 128) * 2;
  size_t sh3 = (size_t)(64 * 512 + 512 * 64 + 16 * 512 + 16 * 64) * 2 + (size_t)(64 + 64 + 16 + 16 + 16) * 4;
  size_t sh4 = (size_t)(32 * 512 + 128 * 512) * 2;

  // gfx1250 has 320KB LDS per WGP; raise dynamic-LDS caps (deterministic, no stream ops)
  hipFuncSetAttribute((const void*)k_qproj,   hipFuncAttributeMaxDynamicSharedMemorySize, (int)sh1);
  hipFuncSetAttribute((const void*)k_qabsorb, hipFuncAttributeMaxDynamicSharedMemorySize, (int)sh2);
  hipFuncSetAttribute((const void*)k_attn,    hipFuncAttributeMaxDynamicSharedMemorySize, (int)sh3);
  hipFuncSetAttribute((const void*)k_outproj, hipFuncAttributeMaxDynamicSharedMemorySize, (int)sh4);

  k_qproj  <<<dim3((H_ * DH_) / 256), 128, sh1, stream>>>(hq, wq, bq, q_bf);
  k_qabsorb<<<dim3(H_),               256, sh2, stream>>>(q_bf, w_kc, qa_bf);
  k_attn   <<<dim3(8, B_),            128, sh3, stream>>>(qa_bf, ckv, ctx_bf);
  k_outproj<<<dim3(H_),               128, sh4, stream>>>(ctx_bf, w_vc, out);
}
